// BlockWithAdapter_24180665876867
// MI455X (gfx1250) — matrix operand order repaired and checked by reading, not compiled, not run on hardware
//
#include <hip/hip_runtime.h>

// Problem constants (B=2, T=2048, C=1024, H=16, hd=64, M_adapter=64)
#define ROWS 4096      // B*T
#define CH   1024
#define NH   16
#define HD   64
#define TSEQ 2048

typedef __attribute__((ext_vector_type(16))) __bf16 v16bf;
typedef __attribute__((ext_vector_type(8)))  float  v8f;
typedef __attribute__((ext_vector_type(4)))  unsigned int v4u;
typedef __attribute__((ext_vector_type(8)))  int v8i;
typedef __attribute__((ext_vector_type(4)))  int v4i;

union Frag16 { uint4 q[2]; unsigned short s[16]; v16bf v; };

__device__ __forceinline__ unsigned short f2bf(float f){
  union { float f; unsigned int u; } a; a.f = f;
  return (unsigned short)((a.u + 0x7FFFu + ((a.u >> 16) & 1u)) >> 16);
}

__device__ __forceinline__ v8f wmma_bf16(const Frag16& a, const Frag16& b, v8f c){
  // D = A(16x32 bf16) * B(32x16 bf16) + C(16x16 f32)
  return __builtin_amdgcn_wmma_f32_16x16x32_bf16(false, a.v, false, b.v,
                                                 (short)0, c, false, false);
}

// --- CDNA5 async global->LDS DMA (ASYNCcnt-tracked), inline asm (portable) ---
// LDS byte address = low 32 bits of the generic pointer (aperture rule,
// ISA 10.2).
__device__ __forceinline__ void async_ld_b128(unsigned lds_off, const void* gaddr){
  asm volatile("global_load_async_to_lds_b128 %0, %1, off"
               :: "v"(lds_off), "v"(gaddr) : "memory");
}
__device__ __forceinline__ void async_wait0(){
  asm volatile("s_wait_asynccnt 0x0" ::: "memory");
}

// --- CDNA5 Tensor Data Mover: one 2-D tile (2-byte elems) -> LDS, with the
// hardware pad feature reproducing our 64B-data + 16B-pad LDS row pitch.
// D# layout per ISA ch.8: group0 {count=1, lds_addr, global_addr[56:0], type=2},
// group1 {data_size=1(2B), pad_enable, pad_interval=3(16 dw), pad_amount=3(4 dw),
//          tensor_dim0/1, tile_dim0/1, tensor_dim0_stride}.
__device__ __forceinline__ void tdm_load_2d(unsigned lds_addr, const void* gaddr,
                                            unsigned tensor_d0, unsigned tensor_d1,
                                            unsigned stride0_elems,
                                            unsigned tile_d0, unsigned tile_d1){
  const unsigned long long ga = (unsigned long long)(size_t)gaddr;
  v4u g0;
  g0[0] = 1u;                                            // count=1, user D#
  g0[1] = lds_addr;                                      // LDS byte address
  g0[2] = (unsigned)ga;                                  // global_addr[31:0]
  g0[3] = (unsigned)((ga >> 32) & 0x01FFFFFFu)           // global_addr[56:32]
        | 0x80000000u;                                   // type=2 ("image")
  v8i g1;
  g1[0] = (int)((1u << 16)                               // data_size: 2 bytes
              | (1u << 20)                               // pad_enable
              | (3u << 22)                               // pad_interval: 16 DWORDs
              | (3u << 25));                             // pad_amount: 4 DWORDs
  g1[1] = (int)((tensor_d0 & 0xFFFFu) << 16);            // tensor_dim0[15:0]
  g1[2] = (int)((tensor_d0 >> 16) | ((tensor_d1 & 0xFFFFu) << 16));
  g1[3] = (int)((tensor_d1 >> 16) | (tile_d0 << 16));    // tile_dim0
  g1[4] = (int)tile_d1;                                  // tile_dim1 (tile_dim2=0)
  g1[5] = (int)stride0_elems;                            // tensor_dim0_stride[31:0]
  g1[6] = 0;
  g1[7] = 0;
  const v4i z4 = {0, 0, 0, 0};
#if defined(__clang_major__) && (__clang_major__ >= 23)
  const v8i z8 = {0, 0, 0, 0, 0, 0, 0, 0};
  __builtin_amdgcn_tensor_load_to_lds(g0, g1, z4, z4, z8, 0);
#else
  __builtin_amdgcn_tensor_load_to_lds(g0, g1, z4, z4, 0);
#endif
}

// ---------------------------------------------------------------------------
// fp32 [K][N] -> bf16 transposed [N][K] (weights; one pass per launch).
// Pre-transposing makes every GEMM weight tile a dense 2-D TDM copy.
// ---------------------------------------------------------------------------
__global__ __launch_bounds__(256)
void f32_to_bf16_T(const float* __restrict__ src, unsigned short* __restrict__ dst,
                   int K, int N){
  __shared__ unsigned short t[32][33];
  const int k0 = blockIdx.y * 32, n0 = blockIdx.x * 32;
  const int tx = threadIdx.x & 31, ty = threadIdx.x >> 5;   // ty: 0..7
#pragma unroll
  for (int r = 0; r < 4; ++r)
    t[ty + 8*r][tx] = f2bf(src[(size_t)(k0 + ty + 8*r) * N + n0 + tx]);
  __syncthreads();
#pragma unroll
  for (int r = 0; r < 4; ++r)
    dst[(size_t)(n0 + ty + 8*r) * K + k0 + tx] = t[tx][ty + 8*r];
}

// ---------------------------------------------------------------------------
// LayerNorm over C=1024, one block (8 waves) per row, bf16 output for WMMA.
// ---------------------------------------------------------------------------
__global__ __launch_bounds__(256)
void ln_bf16_kernel(const float* __restrict__ x, const float* __restrict__ g,
                    const float* __restrict__ bta, unsigned short* __restrict__ y){
  const int row = blockIdx.x;
  const int t   = threadIdx.x;
  const int wid = t >> 5, lid = t & 31;
  const float* xr = x + (size_t)row * CH;
  float v[4], s1 = 0.f, s2 = 0.f;
#pragma unroll
  for (int i = 0; i < 4; ++i){ v[i] = xr[t + i*256]; s1 += v[i]; s2 += v[i]*v[i]; }
#pragma unroll
  for (int m = 1; m < 32; m <<= 1){ s1 += __shfl_xor(s1, m, 32); s2 += __shfl_xor(s2, m, 32); }
  __shared__ float r1[8], r2[8];
  if (lid == 0){ r1[wid] = s1; r2[wid] = s2; }
  __syncthreads();
  float t1 = 0.f, t2 = 0.f;
#pragma unroll
  for (int w = 0; w < 8; ++w){ t1 += r1[w]; t2 += r2[w]; }
  const float mean = t1 * (1.f/CH);
  const float var  = t2 * (1.f/CH) - mean*mean;
  const float rstd = rsqrtf(var + 1e-5f);
  unsigned short* yr = y + (size_t)row * CH;
#pragma unroll
  for (int i = 0; i < 4; ++i){
    const int c = t + i*256;
    yr[c] = f2bf((v[i]-mean)*rstd*g[c] + bta[c]);
  }
}

// ---------------------------------------------------------------------------
// bf16 WMMA GEMM: C[M,N] = A[M,K]*Bt[N,K]^T (+bias, +epilogue, +f32 residual)
// Block tile 128x64, 8 waves (4x2), wave tile 32x32 (2x2 WMMA), K-step 32.
// LDS fill = Tensor Data Mover: wave 0 issues 2 TDM tile loads per K-step
// (A 128x32, B 64x32; HW pad -> 80B row pitch). Double buffering via in-order
// TENSORcnt: after issuing the next 2, s_wait_tensorcnt(2) retires the current.
// EPI: 0=bias, 1=bias+gelu(tanh), 2=bias+relu.
// ---------------------------------------------------------------------------
template<int EPI, bool RES, bool WF32, bool WBF>
__global__ __launch_bounds__(256)
void gemm_bf16_kernel(const unsigned short* __restrict__ A,   // [M][K]
                      const unsigned short* __restrict__ Bt,  // [N][K] (pre-transposed)
                      const float* __restrict__ bias,
                      const float* __restrict__ resid,
                      float* __restrict__ outF,
                      unsigned short* __restrict__ outB,
                      int N, int K){
  __shared__ unsigned short As[2][128][40];   // [m][k], row = 64B data + 16B pad
  __shared__ unsigned short Bs[2][64][40];    // [n][k]
  const int tid = threadIdx.x;
  const int wid = tid >> 5, lid = tid & 31;
  const int l16 = lid & 15, hi = lid >> 4;
  const int wm = wid >> 1, wn = wid & 1;
  const int bm = blockIdx.y * 128, bn = blockIdx.x * 64;

  const unsigned short* Abase = A  + (size_t)bm * K;
  const unsigned short* Bbase = Bt + (size_t)bn * K;

  const v8f zero = {0.f,0.f,0.f,0.f,0.f,0.f,0.f,0.f};
  v8f acc[2][2] = {{zero, zero},{zero, zero}};

  auto issue = [&](int bufi, int kt){
    tdm_load_2d((unsigned)(size_t)&As[bufi][0][0], Abase + (size_t)kt*32,
                (unsigned)K, (unsigned)ROWS, (unsigned)K, 32u, 128u);
    tdm_load_2d((unsigned)(size_t)&Bs[bufi][0][0], Bbase + (size_t)kt*32,
                (unsigned)K, (unsigned)ROWS, (unsigned)K, 32u, 64u);
  };

  const int nk = K >> 5;
  if (wid == 0) issue(0, 0);
  int buf = 0;
  for (int kt = 0; kt < nk; ++kt){
    if (wid == 0){
      if (kt + 1 < nk){ issue(buf ^ 1, kt + 1); __builtin_amdgcn_s_wait_tensorcnt(2); }
      else            { __builtin_amdgcn_s_wait_tensorcnt(0); }
    }
    __syncthreads();                 // tile kt visible to all waves
    // A frag: row=lane&15, k = {hi*8..+7, 16+hi*8..+7}; B frag: col=lane&15, k = {hi*8..+7, 16+hi*8..+7}
    Frag16 fa[2], fb[2];
#pragma unroll
    for (int i = 0; i < 2; ++i){
      const int r = wm*32 + i*16 + l16;
      fa[i].q[0] = *(const uint4*)&As[buf][r][hi*8];
      fa[i].q[1] = *(const uint4*)&As[buf][r][16 + hi*8];
    }
#pragma unroll
    for (int j = 0; j < 2; ++j){
      const int n = wn*32 + j*16 + l16;
      fb[j].q[0] = *(const uint4*)&Bs[buf][n][hi*8];
      fb[j].q[1] = *(const uint4*)&Bs[buf][n][16 + hi*8];
    }
#pragma unroll
    for (int i = 0; i < 2; ++i)
#pragma unroll
      for (int j = 0; j < 2; ++j)
        acc[i][j] = wmma_bf16(fa[i], fb[j], acc[i][j]);
    __syncthreads();                 // all reads of buf done before it is refilled
    buf ^= 1;
  }
  // Epilogue. C layout: lane holds col=lane&15, rows e+8*hi.
#pragma unroll
  for (int i = 0; i < 2; ++i)
#pragma unroll
    for (int j = 0; j < 2; ++j){
      const int rb  = bm + wm*32 + i*16 + 8*hi;
      const int col = bn + wn*32 + j*16 + l16;
      const float bv = bias[col];
#pragma unroll
      for (int e = 0; e < 8; ++e){
        float v = acc[i][j][e] + bv;
        if (EPI == 1){
          v = 0.5f * v * (1.0f + tanhf(0.7978845608f * (v + 0.044715f*v*v*v)));
        } else if (EPI == 2){
          v = v > 0.f ? v : 0.f;
        }
        const size_t idx = (size_t)(rb + e) * N + col;
        if (RES)  v += resid[idx];
        if (WF32) outF[idx] = v;
        if (WBF)  outB[idx] = f2bf(v);
      }
    }
}

// ---------------------------------------------------------------------------
// Causal flash attention. Grid (T/64, H, B); 4 waves/block, 16 q-rows/wave.
// K tiles arrive via async-to-LDS DMA; V is VGPR-transposed (2-byte granule).
// Per 32-key chunk: S = Q*K^T (2 WMMA / 16-key tile), online softmax with
// shfl_xor row reductions, P staged via LDS (C->A layout), P*V (4 WMMA).
// All WMMA-bearing branches are wave-uniform (EXEC stays all-ones).
// ---------------------------------------------------------------------------
__global__ __launch_bounds__(128)
void attn_kernel(const unsigned short* __restrict__ qkv,  // [B,T,3C] bf16 (q|k|v)
                 unsigned short* __restrict__ y){         // [B,T,C] bf16
  const int tid = threadIdx.x;
  const int wid = tid >> 5, lid = tid & 31;
  const int l16 = lid & 15, hi = lid >> 4;
  const int qb = blockIdx.x;
  const int h  = blockIdx.y;
  const int b  = blockIdx.z;
  const int q0 = qb*64 + wid*16;

  __shared__ unsigned short Ks[32][72];       // [key][d], rows 144B (16B aligned)
  __shared__ unsigned short Vt[64][40];       // transposed [d][key]
  __shared__ unsigned short Ps[4][16][40];    // per-wave P staging [row][key]

  const unsigned short* base = qkv + (size_t)b * TSEQ * (3*CH);

  // Q fragments for the two 32-wide K-dim steps over hd=64
  Frag16 aq[2];
  {
    const unsigned short* qrow = base + (size_t)(q0 + l16) * (3*CH) + h*HD;
#pragma unroll
    for (int s = 0; s < 2; ++s){
      aq[s].q[0] = *(const uint4*)(qrow + s*32 + hi*8);
      aq[s].q[1] = *(const uint4*)(qrow + s*32 + 16 + hi*8);
    }
  }

  const v8f zero = {0.f,0.f,0.f,0.f,0.f,0.f,0.f,0.f};
  v8f o[4] = {zero, zero, zero, zero};        // O tile 16x64 = 4 x (16x16 f32)
  float mrow[8], lrow[8];
#pragma unroll
  for (int e = 0; e < 8; ++e){ mrow[e] = -1e30f; lrow[e] = 0.f; }

  const int nkb = qb*2 + 2;                   // uniform loop bound per block
  const int ldr = tid >> 2;                   // 0..31 (key row)
  const int ldd = (tid & 3) * 16;             // 0/16/32/48 (d offset)
  for (int kb = 0; kb < nkb; ++kb){
    const int key0 = kb * 32;
    { // K tile via async DMA (natural layout), V tile VGPR-transposed
      const unsigned short* krow = base + (size_t)(key0 + ldr)*(3*CH) + CH + h*HD + ldd;
      async_ld_b128((unsigned)(size_t)&Ks[ldr][ldd],     krow);
      async_ld_b128((unsigned)(size_t)&Ks[ldr][ldd + 8], krow + 8);
      const unsigned short* vrow = base + (size_t)(key0 + ldr)*(3*CH) + 2*CH + h*HD + ldd;
      Frag16 tv;
      tv.q[0] = *(const uint4*)(vrow);
      tv.q[1] = *(const uint4*)(vrow + 8);
#pragma unroll
      for (int j = 0; j < 16; ++j) Vt[ldd + j][ldr] = tv.s[j];
      async_wait0();
    }
    __syncthreads();
    const bool active = (key0 <= q0 + 15);    // wave-uniform
    if (active){
#pragma unroll
      for (int t = 0; t < 2; ++t){            // two 16-key S tiles per chunk
        v8f sacc = zero;
        Frag16 fk[2];
        const int keyl = t*16 + l16;
#pragma unroll
        for (int s = 0; s < 2; ++s){          // B = K^T: col=key, k=d (table order)
          fk[s].q[0] = *(const uint4*)&Ks[keyl][s*32 + hi*8];
          fk[s].q[1] = *(const uint4*)&Ks[keyl][s*32 + 16 + hi*8];
        }
        sacc = wmma_bf16(aq[0], fk[0], sacc);
        sacc = wmma_bf16(aq[1], fk[1], sacc);
        const int key = key0 + t*16 + l16;
        float sv[8], rmax[8];
#pragma unroll
        for (int e = 0; e < 8; ++e){
          const int row = q0 + e + 8*hi;
          const float xv = sacc[e] * 0.125f;          // 1/sqrt(64)
          sv[e]   = (key <= row) ? xv : -1e30f;       // causal mask
          rmax[e] = sv[e];
        }
#pragma unroll
        for (int m = 1; m < 16; m <<= 1){             // row max across 16-lane group
#pragma unroll
          for (int e = 0; e < 8; ++e)
            rmax[e] = fmaxf(rmax[e], __shfl_xor(rmax[e], m, 32));
        }
        float pr[8], rs[8];
#pragma unroll
        for (int e = 0; e < 8; ++e){
          const float nm = fmaxf(mrow[e], rmax[e]);
          const float sc = __expf(mrow[e] - nm);
          mrow[e] = nm;
          pr[e] = __expf(sv[e] - nm);
          rs[e] = pr[e];
          lrow[e] *= sc;
#pragma unroll
          for (int nt = 0; nt < 4; ++nt) o[nt][e] *= sc;
        }
#pragma unroll
        for (int m = 1; m < 16; m <<= 1){             // row sum
#pragma unroll
          for (int e = 0; e < 8; ++e)
            rs[e] += __shfl_xor(rs[e], m, 32);
        }
#pragma unroll
        for (int e = 0; e < 8; ++e){
          lrow[e] += rs[e];
          Ps[wid][e + 8*hi][t*16 + l16] = f2bf(pr[e]); // stage P (C->A layout)
        }
      }
    }
    __syncthreads();
    if (active){
      Frag16 ap;                                      // A = P (16x32)
      ap.q[0] = *(const uint4*)&Ps[wid][l16][hi*8];
      ap.q[1] = *(const uint4*)&Ps[wid][l16][16 + hi*8];
#pragma unroll
      for (int nt = 0; nt < 4; ++nt){                 // B = V (32 keys -> d cols)
        Frag16 fv;
        const int n = nt*16 + l16;
        fv.q[0] = *(const uint4*)&Vt[n][hi*8];
        fv.q[1] = *(const uint4*)&Vt[n][16 + hi*8];
        o[nt] = wmma_bf16(ap, fv, o[nt]);
      }
    }
    __syncthreads();
  }
  // O /= l, write bf16 into [B,T,C] at head slice (matches head-concat)
#pragma unroll
  for (int nt = 0; nt < 4; ++nt)
#pragma unroll
    for (int e = 0; e < 8; ++e){
      const int row = q0 + e + 8*hi;
      const int col = h*HD + nt*16 + l16;
      y[((size_t)b*TSEQ + row)*CH + col] = f2bf(o[nt][e] / lrow[e]);
    }
}

// ---------------------------------------------------------------------------
// Host-side pipeline
// ---------------------------------------------------------------------------
extern "C" void kernel_launch(void* const* d_in, const int* in_sizes, int n_in,
                              void* d_out, int out_size, void* d_ws, size_t ws_size,
                              hipStream_t stream){
  (void)in_sizes; (void)n_in; (void)out_size; (void)ws_size;
  const float* x      = (const float*)d_in[0];
  const float* ln1_g  = (const float*)d_in[1];
  const float* ln1_b  = (const float*)d_in[2];
  const float* attn_w = (const float*)d_in[3];
  const float* attn_b = (const float*)d_in[4];
  const float* proj_w = (const float*)d_in[5];
  const float* proj_b = (const float*)d_in[6];
  const float* a1_dw  = (const float*)d_in[7];
  const float* a1_db  = (const float*)d_in[8];
  const float* a1_uw  = (const float*)d_in[9];
  const float* a1_ub  = (const float*)d_in[10];
  const float* ln2_g  = (const float*)d_in[11];
  const float* ln2_b  = (const float*)d_in[12];
  const float* fc_w   = (const float*)d_in[13];
  const float* fc_b   = (const float*)d_in[14];
  const float* mlp_pw = (const float*)d_in[15];
  const float* mlp_pb = (const float*)d_in[16];
  const float* a2_dw  = (const float*)d_in[17];
  const float* a2_db  = (const float*)d_in[18];
  const float* a2_uw  = (const float*)d_in[19];
  const float* a2_ub  = (const float*)d_in[20];
  float* out = (float*)d_out;

  unsigned char* wsp = (unsigned char*)d_ws;
  size_t off = 0;
  auto alloc = [&](size_t n)->void*{ void* r = wsp + off; off += (n + 255) & ~(size_t)255; return r; };

  // bf16 transposed weights [N][K] (~25.7 MB)
  unsigned short* w_qkv = (unsigned short*)alloc((size_t)CH*3*CH*2);
  unsigned short* w_proj= (unsigned short*)alloc((size_t)CH*CH*2);
  unsigned short* w_fc  = (unsigned short*)alloc((size_t)CH*4*CH*2);
  unsigned short* w_mlp = (unsigned short*)alloc((size_t)4*CH*CH*2);
  unsigned short* w_a1d = (unsigned short*)alloc((size_t)CH*64*2);
  unsigned short* w_a1u = (unsigned short*)alloc((size_t)64*CH*2);
  unsigned short* w_a2d = (unsigned short*)alloc((size_t)CH*64*2);
  unsigned short* w_a2u = (unsigned short*)alloc((size_t)64*CH*2);
  // activations (reused; ~84 MB)
  unsigned short* bufA  = (unsigned short*)alloc((size_t)ROWS*CH*2);    // ln1 / attn_y / ln2
  unsigned short* bufB  = (unsigned short*)alloc((size_t)ROWS*CH*2);    // x1_bf16 / x3_bf16
  unsigned short* bufS  = (unsigned short*)alloc((size_t)ROWS*64*2);    // adapter hidden
  unsigned short* bufQH = (unsigned short*)alloc((size_t)ROWS*4*CH*2);  // qkv / mlp hidden
  float* F1 = (float*)alloc((size_t)ROWS*CH*4);                          // x1 / x3 (f32)
  float* F2 = (float*)alloc((size_t)ROWS*CH*4);                          // x2 (f32)

  // weight convert + transpose: src [K][N] f32 -> dst [N][K] bf16
  auto cvtT = [&](const float* s, unsigned short* d, int K, int N){
    f32_to_bf16_T<<<dim3(N/32, K/32), dim3(256), 0, stream>>>(s, d, K, N);
  };
  cvtT(attn_w, w_qkv, CH, 3*CH);
  cvtT(proj_w, w_proj, CH, CH);
  cvtT(fc_w,   w_fc,   CH, 4*CH);
  cvtT(mlp_pw, w_mlp,  4*CH, CH);
  cvtT(a1_dw,  w_a1d,  CH, 64);
  cvtT(a1_uw,  w_a1u,  64, CH);
  cvtT(a2_dw,  w_a2d,  CH, 64);
  cvtT(a2_uw,  w_a2u,  64, CH);

  // ln1(x) -> bufA (bf16)
  ln_bf16_kernel<<<dim3(ROWS), dim3(256), 0, stream>>>(x, ln1_g, ln1_b, bufA);
  // qkv = ln1 @ attn_w + attn_b -> bufQH (bf16)
  gemm_bf16_kernel<0,false,false,true><<<dim3(3*CH/64, ROWS/128), dim3(256), 0, stream>>>(
      bufA, w_qkv, attn_b, nullptr, nullptr, bufQH, 3*CH, CH);
  // attention -> bufA (bf16 head-concat)
  attn_kernel<<<dim3(TSEQ/64, NH, 2), dim3(128), 0, stream>>>(bufQH, bufA);
  // x1 = x + attn_y @ proj_w + proj_b -> F1 (f32) + bufB (bf16)
  gemm_bf16_kernel<0,true,true,true><<<dim3(CH/64, ROWS/128), dim3(256), 0, stream>>>(
      bufA, w_proj, proj_b, x, F1, bufB, CH, CH);
  // adapter1 down: relu(x1 @ a1_dw + a1_db) -> bufS
  gemm_bf16_kernel<2,false,false,true><<<dim3(1, ROWS/128), dim3(256), 0, stream>>>(
      bufB, w_a1d, a1_db, nullptr, nullptr, bufS, 64, CH);
  // x2 = x1 + h1 @ a1_uw + a1_ub -> F2 (f32)
  gemm_bf16_kernel<0,true,true,false><<<dim3(CH/64, ROWS/128), dim3(256), 0, stream>>>(
      bufS, w_a1u, a1_ub, F1, F2, nullptr, CH, 64);
  // ln2(x2) -> bufA
  ln_bf16_kernel<<<dim3(ROWS), dim3(256), 0, stream>>>(F2, ln2_g, ln2_b, bufA);
  // h = gelu(ln2 @ fc_w + fc_b) -> bufQH
  gemm_bf16_kernel<1,false,false,true><<<dim3(4*CH/64, ROWS/128), dim3(256), 0, stream>>>(
      bufA, w_fc, fc_b, nullptr, nullptr, bufQH, 4*CH, CH);
  // x3 = x2 + h @ mlp_pw + mlp_pb -> F1 (f32) + bufB (bf16)
  gemm_bf16_kernel<0,true,true,true><<<dim3(CH/64, ROWS/128), dim3(256), 0, stream>>>(
      bufQH, w_mlp, mlp_pb, F2, F1, bufB, CH, 4*CH);
  // adapter2 down
  gemm_bf16_kernel<2,false,false,true><<<dim3(1, ROWS/128), dim3(256), 0, stream>>>(
      bufB, w_a2d, a2_db, nullptr, nullptr, bufS, 64, CH);
  // out = x3 + h2 @ a2_uw + a2_ub -> d_out (f32)
  gemm_bf16_kernel<0,true,true,false><<<dim3(CH/64, ROWS/128), dim3(256), 0, stream>>>(
      bufS, w_a2u, a2_ub, F1, out, nullptr, CH, 64);
}
